// MIGT_90787018703328
// MI455X (gfx1250) — compile-verified, matrix operand order checked by reading, not run on hardware
//
#include <hip/hip_runtime.h>
#include <hip/hip_bf16.h>
#include <math.h>

// ---------------------------------------------------------------------------
// Shapes (compile-time constants from the reference)
// ---------------------------------------------------------------------------
#define BATCH      2
#define N_BLOCKS   20
#define BLOCK_SZ   64
#define D_MODEL    768
#define N_HEAD     12
#define D_HEAD     64          // 768 / 12
#define D_INNER    3072
#define TOK_STREAM (BATCH * N_BLOCKS * BLOCK_SZ)   // 2560 tokens per stream
#define TOK_ALL    (2 * TOK_STREAM)                 // 5120
#define QKV_N      (3 * D_MODEL)                    // 2304
#define Y_ELEMS    (TOK_STREAM * D_MODEL)           // 1966080 floats per y

typedef __attribute__((ext_vector_type(16))) _Float16 v16h;
typedef __attribute__((ext_vector_type(8)))  float    v8f;
typedef __attribute__((ext_vector_type(8)))  _Float16 hvec8;

// ---------------------------------------------------------------------------
// CDNA5 async global->LDS copy (ASYNCcnt-tracked, no VGPR round trip).
// VDST is the wave-relative LDS byte offset = low 32 bits of a generic
// pointer to __shared__ (flat aperture rules, ISA 10.2).
// ---------------------------------------------------------------------------
__device__ __forceinline__ void async_copy_b128(void* lds_ptr, const void* gptr) {
  const unsigned lds_off = (unsigned)(unsigned long long)lds_ptr;
  asm volatile("global_load_async_to_lds_b128 %0, %1, off"
               :: "v"(lds_off), "v"(gptr) : "memory");
}
__device__ __forceinline__ void wait_async_zero() {
  asm volatile("s_wait_asynccnt 0x0" ::: "memory");
}

// ---------------------------------------------------------------------------
// WMMA fragment loader.  Layout per CDNA5 ISA (16-bit A 16x32 / B-as-Bt):
//   lane 0-15  : row m = lane,    halfs 0..7 -> k=0..7,  halfs 8..15 -> k=16..23
//   lane 16-31 : row m = lane-16, halfs 0..7 -> k=8..15, halfs 8..15 -> k=24..31
// `base` points at (row 0, k 0) of a row-major tile with leading dim `ld`.
// ---------------------------------------------------------------------------
__device__ __forceinline__ v16h frag16(const _Float16* base, int ld, int lane) {
  const int m  = lane & 15;
  const int kb = (lane >> 4) << 3;           // 0 or 8
  const hvec8 lo = *(const hvec8*)(base + m * ld + kb);
  const hvec8 hi = *(const hvec8*)(base + m * ld + kb + 16);
  v16h r;
#pragma unroll
  for (int t = 0; t < 8; ++t) { r[t] = lo[t]; r[8 + t] = hi[t]; }
  return r;
}

__device__ __forceinline__ v8f wmma_f16(v16h a, v16h b, v8f c) {
  return __builtin_amdgcn_wmma_f32_16x16x32_f16(false, a, false, b,
                                                (short)0, c, false, false);
}

__device__ __forceinline__ float gelu_exact(float x) {
  return 0.5f * x * (1.0f + erff(x * 0.70710678118654752440f));
}

// ---------------------------------------------------------------------------
// fp32 -> fp16 conversion
// ---------------------------------------------------------------------------
__global__ void cvt_f2h_kernel(const float* __restrict__ src,
                               _Float16* __restrict__ dst, int n) {
  int i = blockIdx.x * blockDim.x + threadIdx.x;
  if (i < n) dst[i] = (_Float16)src[i];
}

// ---------------------------------------------------------------------------
// LayerNorm over D_MODEL, fp32 in -> fp16 out. One block per token, 256 thr.
// ---------------------------------------------------------------------------
__global__ __launch_bounds__(256)
void ln_kernel(const float* __restrict__ x, const float* __restrict__ g,
               const float* __restrict__ bta, _Float16* __restrict__ out) {
  const int t = blockIdx.x;
  const float* row = x + (size_t)t * D_MODEL;
  float s = 0.f, s2 = 0.f;
  for (int j = threadIdx.x; j < D_MODEL; j += 256) {
    float v = row[j]; s += v; s2 += v * v;
  }
#pragma unroll
  for (int msk = 16; msk > 0; msk >>= 1) {
    s  += __shfl_xor(s,  msk, 32);
    s2 += __shfl_xor(s2, msk, 32);
  }
  __shared__ float sh[16];
  __shared__ float mv[2];
  const int wv = threadIdx.x >> 5;
  if ((threadIdx.x & 31) == 0) { sh[wv] = s; sh[8 + wv] = s2; }
  __syncthreads();
  if (threadIdx.x == 0) {
    float ts = 0.f, ts2 = 0.f;
#pragma unroll
    for (int k = 0; k < 8; ++k) { ts += sh[k]; ts2 += sh[8 + k]; }
    float mean = ts * (1.0f / D_MODEL);
    float var  = ts2 * (1.0f / D_MODEL) - mean * mean;
    mv[0] = mean; mv[1] = rsqrtf(var + 1e-5f);
  }
  __syncthreads();
  const float mean = mv[0], rstd = mv[1];
  for (int j = threadIdx.x; j < D_MODEL; j += 256) {
    out[(size_t)t * D_MODEL + j] =
        (_Float16)(((row[j] - mean) * rstd) * g[j] + bta[j]);
  }
}

// ---------------------------------------------------------------------------
// Tiled WMMA GEMM:  C[M,N] = A[M,K](f16) * B[K,N](f16) + bias
//   op 0: store f16          op 1: GELU, store f16
//   op 2: +residual(f32), store f32
// 64x64 tile / block, 256 threads (8 waves), each wave: 2 x 16x16 subtiles.
// A tile staged with CDNA5 async global->LDS copies; B staged transposed
// through VGPRs (permutation, cannot async-copy).
// ---------------------------------------------------------------------------
#define G_LDA 40   // 32 + 8 pad (halfs)
#define G_LDB 40

__global__ __launch_bounds__(256)
void gemm_wmma_kernel(const _Float16* __restrict__ A, int lda,
                      const _Float16* __restrict__ B, int ldb,
                      const float* __restrict__ bias,
                      const float* __restrict__ residual,
                      float* __restrict__ outF,
                      _Float16* __restrict__ outH,
                      int M, int N, int K, int op) {
  __shared__ _Float16 As[64 * G_LDA];
  __shared__ _Float16 Bs[64 * G_LDB];   // transposed: [n][k]

  const int m0   = blockIdx.y * 64;
  const int n0   = blockIdx.x * 64;
  const int tid  = threadIdx.x;
  const int lane = tid & 31;
  const int wv   = tid >> 5;
  const int sr   = wv >> 1;             // subtile row 0..3
  const int sc0  = (wv & 1) * 2;        // subtile cols {sc0, sc0+1}

  v8f acc0 = {}; v8f acc1 = {};

  const int ar = tid >> 2;              // A stage: 64 rows, 4 thr/row
  const int ac = (tid & 3) * 8;
  const int br = tid >> 3;              // B stage: 32 rows, 8 thr/row
  const int bc = (tid & 7) * 8;

  _Float16* lds_a = &As[ar * G_LDA + ac];
  const _Float16* gA = A + (size_t)(m0 + ar) * lda + ac;

  for (int k0 = 0; k0 < K; k0 += 32) {
    __syncthreads();
    // stage A (64x32) row-major: async global->LDS, no VGPR round trip
    async_copy_b128(lds_a, gA + k0);
    // stage B (32x64) transposed into Bs[n][k]
    {
      const _Float16* bp = B + (size_t)(k0 + br) * ldb + n0 + bc;
      if (k0 + 32 < K)
        __builtin_prefetch(bp + (size_t)32 * ldb, 0, 1);
      const hvec8 vb = *(const hvec8*)bp;
#pragma unroll
      for (int t = 0; t < 8; ++t) Bs[(bc + t) * G_LDB + br] = vb[t];
    }
    wait_async_zero();
    __syncthreads();

    const v16h af  = frag16(&As[(16 * sr) * G_LDA], G_LDA, lane);
    const v16h bf0 = frag16(&Bs[(16 * sc0) * G_LDB], G_LDB, lane);
    const v16h bf1 = frag16(&Bs[(16 * (sc0 + 1)) * G_LDB], G_LDB, lane);
    acc0 = wmma_f16(af, bf0, acc0);
    acc1 = wmma_f16(af, bf1, acc1);
  }

  // Epilogue. C layout: lane holds col (lane&15); VGPR r -> row r + 8*(lane>>4)
  const int ncol   = lane & 15;
  const int halfid = lane >> 4;
#pragma unroll
  for (int u = 0; u < 2; ++u) {
    const v8f acc = u ? acc1 : acc0;
    const int n   = n0 + 16 * (sc0 + u) + ncol;
    const float bv = bias ? bias[n] : 0.f;
    const int mb  = m0 + 16 * sr + 8 * halfid;
#pragma unroll
    for (int r = 0; r < 8; ++r) {
      float v = acc[r] + bv;
      const size_t idx = (size_t)(mb + r) * N + n;
      if (op == 2)      outF[idx] = v + residual[idx];
      else if (op == 1) outH[idx] = (_Float16)gelu_exact(v);
      else              outH[idx] = (_Float16)v;
    }
  }
}

// ---------------------------------------------------------------------------
// Flash-style block attention with WMMA.
// grid = (N_BLOCKS, N_HEAD, 2*BATCH), block = 128 (4 waves, 16 q-rows each).
// QKV layout: [token, 2304] f16 with v|q|k at cols 0|768|1536, head h at +64h.
// Keys/values: blocks j<i from stream 0, diagonal block from own stream.
// Q/K tiles staged with async global->LDS; V staged transposed via VGPRs.
// 1/sqrt(Dh) applied to WMMA score outputs (exact power of two).
// ---------------------------------------------------------------------------
#define A_LD 72   // 64 + 8 pad (halfs)

__global__ __launch_bounds__(128)
void attn_wmma_kernel(const _Float16* __restrict__ QKV,
                      _Float16* __restrict__ out) {
  __shared__ _Float16 Qs[64 * A_LD];
  __shared__ _Float16 Ks[64 * A_LD];   // row-major [key][d]  (== Kt for QK^T)
  __shared__ _Float16 Vt[64 * A_LD];   // transposed [d][key]
  __shared__ _Float16 Ps[64 * A_LD];   // P tile, fp16

  const int ib = blockIdx.x;                 // query block index
  const int hh = blockIdx.y;                 // head
  const int e  = blockIdx.z >> 1;            // stream
  const int bb = blockIdx.z & 1;             // batch
  const int tid  = threadIdx.x;
  const int lane = tid & 31;
  const int wv   = tid >> 5;                 // 0..3, rows 16*wv..16*wv+15
  const int ncol   = lane & 15;
  const int halfid = lane >> 4;

  const size_t qrow0 = (size_t)e * TOK_STREAM + (size_t)bb * (N_BLOCKS * BLOCK_SZ)
                     + (size_t)ib * BLOCK_SZ;
  const int hc = hh * D_HEAD;

  // stage Q (plain copy -> async)
#pragma unroll
  for (int it = 0; it < 4; ++it) {
    const int idx = tid + it * 128;
    const int r = idx >> 3, c = (idx & 7) * 8;
    async_copy_b128(&Qs[r * A_LD + c],
                    &QKV[(qrow0 + r) * QKV_N + D_MODEL + hc + c]);
  }

  float mrow[8], lrow[8];
  v8f o[4];
#pragma unroll
  for (int r = 0; r < 8; ++r) { mrow[r] = -3.0e38f; lrow[r] = 0.f; }
#pragma unroll
  for (int sc = 0; sc < 4; ++sc) o[sc] = v8f{};
  wait_async_zero();
  __syncthreads();

  for (int j = 0; j <= ib; ++j) {
    const size_t krow0 = (j < ib)
        ? ((size_t)bb * (N_BLOCKS * BLOCK_SZ) + (size_t)j * BLOCK_SZ)     // stream 0 past
        : qrow0;                                                           // own diagonal
    // stage K (async straight copy) and V (transposed via VGPRs)
#pragma unroll
    for (int it = 0; it < 4; ++it) {
      const int idx = tid + it * 128;
      const int r = idx >> 3, c = (idx & 7) * 8;
      async_copy_b128(&Ks[r * A_LD + c],
                      &QKV[(krow0 + r) * QKV_N + 2 * D_MODEL + hc + c]);
      const hvec8 vv = *(const hvec8*)(&QKV[(krow0 + r) * QKV_N + hc + c]);
#pragma unroll
      for (int t = 0; t < 8; ++t) Vt[(c + t) * A_LD + r] = vv[t];
    }
    wait_async_zero();
    __syncthreads();

    // S = (1/8) * Q * K^T  (16 x 64 strip per wave)
    v8f s[4];
#pragma unroll
    for (int sc = 0; sc < 4; ++sc) {
      v8f a = {};
#pragma unroll
      for (int kk = 0; kk < 64; kk += 32) {
        const v16h af = frag16(&Qs[(16 * wv) * A_LD + kk], A_LD, lane);
        const v16h bf = frag16(&Ks[(16 * sc) * A_LD + kk], A_LD, lane);
        a = wmma_f16(af, bf, a);
      }
#pragma unroll
      for (int r = 0; r < 8; ++r) a[r] *= 0.125f;   // 1/sqrt(Dh)
      s[sc] = a;
    }

    // online softmax: row max across the 64 keys of this block
    float mloc[8];
#pragma unroll
    for (int r = 0; r < 8; ++r)
      mloc[r] = fmaxf(fmaxf(s[0][r], s[1][r]), fmaxf(s[2][r], s[3][r]));
#pragma unroll
    for (int msk = 1; msk < 16; msk <<= 1) {
#pragma unroll
      for (int r = 0; r < 8; ++r)
        mloc[r] = fmaxf(mloc[r], __shfl_xor(mloc[r], msk, 32));
    }
    float fac[8];
#pragma unroll
    for (int r = 0; r < 8; ++r) {
      const float mn = fmaxf(mrow[r], mloc[r]);
      fac[r] = expf(mrow[r] - mn);
      mrow[r] = mn;
      lrow[r] *= fac[r];
    }
#pragma unroll
    for (int sc = 0; sc < 4; ++sc)
#pragma unroll
      for (int r = 0; r < 8; ++r) o[sc][r] *= fac[r];

    // P = exp(S - m); write fp16 tile; accumulate row sums
    float ps[8];
#pragma unroll
    for (int r = 0; r < 8; ++r) ps[r] = 0.f;
#pragma unroll
    for (int sc = 0; sc < 4; ++sc) {
#pragma unroll
      for (int r = 0; r < 8; ++r) {
        const float p = expf(s[sc][r] - mrow[r]);
        ps[r] += p;
        Ps[(16 * wv + 8 * halfid + r) * A_LD + 16 * sc + ncol] = (_Float16)p;
      }
    }
#pragma unroll
    for (int msk = 1; msk < 16; msk <<= 1) {
#pragma unroll
      for (int r = 0; r < 8; ++r) ps[r] += __shfl_xor(ps[r], msk, 32);
    }
#pragma unroll
    for (int r = 0; r < 8; ++r) lrow[r] += ps[r];

    // O += P * V   (same-wave DS write->read is in order)
#pragma unroll
    for (int sc = 0; sc < 4; ++sc) {
#pragma unroll
      for (int kk = 0; kk < 64; kk += 32) {
        const v16h af = frag16(&Ps[(16 * wv) * A_LD + kk], A_LD, lane);
        const v16h bf = frag16(&Vt[(16 * sc) * A_LD + kk], A_LD, lane);
        o[sc] = wmma_f16(af, bf, o[sc]);
      }
    }
    __syncthreads();   // protect Ks/Vt before next block restages
  }

  // finalize: out[token, h*64 + d] = O / l   (merged-head layout)
#pragma unroll
  for (int sc = 0; sc < 4; ++sc) {
#pragma unroll
    for (int r = 0; r < 8; ++r) {
      const int rowabs = 16 * wv + 8 * halfid + r;
      const int d = 16 * sc + ncol;
      out[(qrow0 + rowabs) * D_MODEL + hc + d] = (_Float16)(o[sc][r] / lrow[r]);
    }
  }
}

// ---------------------------------------------------------------------------
// pose_proj (N=7) + quaternion normalize.  One block per token, 7 waves.
// ---------------------------------------------------------------------------
__global__ __launch_bounds__(224)
void pose_proj_kernel(const _Float16* __restrict__ H,   // [2560 x 1536]
                      const float* __restrict__ W,      // [1536 x 7]
                      const float* __restrict__ bias,   // [7]
                      float* __restrict__ out) {        // [2560 x 7]
  const int t = blockIdx.x;
  const int col  = threadIdx.x >> 5;   // 0..6
  const int lane = threadIdx.x & 31;
  float s = 0.f;
  for (int k = lane; k < 2 * D_MODEL; k += 32)
    s += (float)H[(size_t)t * (2 * D_MODEL) + k] * W[(size_t)k * 7 + col];
#pragma unroll
  for (int msk = 16; msk > 0; msk >>= 1) s += __shfl_xor(s, msk, 32);
  __shared__ float hv[7];
  if (lane == 0) hv[col] = s + bias[col];
  __syncthreads();
  if (threadIdx.x == 0) {
    float q0 = hv[3], q1 = hv[4], q2 = hv[5], q3 = hv[6];
    const float inv = rsqrtf(q0 * q0 + q1 * q1 + q2 * q2 + q3 * q3);
    q0 *= inv; q1 *= inv; q2 *= inv; q3 *= inv;
    if (q0 < 0.f) { q0 = -q0; q1 = -q1; q2 = -q2; q3 = -q3; }
    float* o = out + (size_t)t * 7;
    o[0] = hv[0]; o[1] = hv[1]; o[2] = hv[2];
    o[3] = q0; o[4] = q1; o[5] = q2; o[6] = q3;
  }
}

// ---------------------------------------------------------------------------
// Host-side orchestration
// ---------------------------------------------------------------------------
extern "C" void kernel_launch(void* const* d_in, const int* in_sizes, int n_in,
                              void* d_out, int out_size, void* d_ws, size_t ws_size,
                              hipStream_t stream) {
  (void)in_sizes; (void)n_in; (void)out_size; (void)ws_size;

  const float* x0      = (const float*)d_in[0];
  const float* x1      = (const float*)d_in[1];
  const float* ln1_g   = (const float*)d_in[2];
  const float* ln1_b   = (const float*)d_in[3];
  const float* attn_w  = (const float*)d_in[4];
  const float* attn_b  = (const float*)d_in[5];
  const float* proj_w  = (const float*)d_in[6];
  const float* proj_b  = (const float*)d_in[7];
  const float* ln2_g   = (const float*)d_in[8];
  const float* ln2_b   = (const float*)d_in[9];
  const float* fc_w    = (const float*)d_in[10];
  const float* fc_b    = (const float*)d_in[11];
  const float* mproj_w = (const float*)d_in[12];
  const float* mproj_b = (const float*)d_in[13];
  const float* pfc_w   = (const float*)d_in[14];
  const float* pfc_b   = (const float*)d_in[15];
  const float* ppr_w   = (const float*)d_in[16];
  const float* ppr_b   = (const float*)d_in[17];
  float* outp = (float*)d_out;

  // workspace carve-up (256B aligned)
  char* ws = (char*)d_ws;
  size_t off = 0;
  auto carve = [&](size_t bytes) -> char* {
    char* p = ws + off;
    off = (off + bytes + 255) & ~(size_t)255;
    return p;
  };
  _Float16* Wqkv   = (_Float16*)carve((size_t)D_MODEL * QKV_N * 2);
  _Float16* Wproj  = (_Float16*)carve((size_t)D_MODEL * D_MODEL * 2);
  _Float16* Wfc    = (_Float16*)carve((size_t)D_MODEL * D_INNER * 2);
  _Float16* Wmproj = (_Float16*)carve((size_t)D_INNER * D_MODEL * 2);
  _Float16* Wpfc   = (_Float16*)carve((size_t)D_MODEL * 2 * D_MODEL * 2);
  _Float16* Ah     = (_Float16*)carve((size_t)TOK_ALL * D_MODEL * 2);
  _Float16* QKVh   = (_Float16*)carve((size_t)TOK_ALL * QKV_N * 2);
  _Float16* Atth   = (_Float16*)carve((size_t)TOK_ALL * D_MODEL * 2);
  float*    Xmid   = (float*)   carve((size_t)TOK_ALL * D_MODEL * 4);
  _Float16* Mh     = (_Float16*)carve((size_t)TOK_ALL * D_MODEL * 2);
  _Float16* Hh     = (_Float16*)carve((size_t)TOK_ALL * D_INNER * 2);
  _Float16* PoseH  = (_Float16*)carve((size_t)TOK_STREAM * 2 * D_MODEL * 2);

  auto cvt = [&](const float* s, _Float16* d, int n) {
    cvt_f2h_kernel<<<(n + 255) / 256, 256, 0, stream>>>(s, d, n);
  };

  // 1) weights fp32 -> fp16
  cvt(attn_w,  Wqkv,   D_MODEL * QKV_N);
  cvt(proj_w,  Wproj,  D_MODEL * D_MODEL);
  cvt(fc_w,    Wfc,    D_MODEL * D_INNER);
  cvt(mproj_w, Wmproj, D_INNER * D_MODEL);
  cvt(pfc_w,   Wpfc,   D_MODEL * 2 * D_MODEL);

  // 2) LN1 on both streams -> Ah (f16)
  ln_kernel<<<TOK_STREAM, 256, 0, stream>>>(x0, ln1_g, ln1_b, Ah);
  ln_kernel<<<TOK_STREAM, 256, 0, stream>>>(x1, ln1_g, ln1_b,
                                            Ah + (size_t)TOK_STREAM * D_MODEL);

  // 3) QKV GEMM: [5120x768] x [768x2304]
  gemm_wmma_kernel<<<dim3(QKV_N / 64, TOK_ALL / 64), 256, 0, stream>>>(
      Ah, D_MODEL, Wqkv, QKV_N, attn_b, nullptr, nullptr, QKVh,
      TOK_ALL, QKV_N, D_MODEL, 0);

  // 4) multi-end block attention
  attn_wmma_kernel<<<dim3(N_BLOCKS, N_HEAD, 2 * BATCH), 128, 0, stream>>>(QKVh, Atth);

  // 5) proj + residual (per stream, residual = original x)
  for (int e = 0; e < 2; ++e) {
    gemm_wmma_kernel<<<dim3(D_MODEL / 64, TOK_STREAM / 64), 256, 0, stream>>>(
        Atth + (size_t)e * TOK_STREAM * D_MODEL, D_MODEL,
        Wproj, D_MODEL, proj_b,
        e ? x1 : x0,
        Xmid + (size_t)e * TOK_STREAM * D_MODEL, nullptr,
        TOK_STREAM, D_MODEL, D_MODEL, 2);
  }

  // 6) LN2 -> Mh (f16)
  ln_kernel<<<TOK_ALL, 256, 0, stream>>>(Xmid, ln2_g, ln2_b, Mh);

  // 7) MLP fc + GELU -> Hh (f16)
  gemm_wmma_kernel<<<dim3(D_INNER / 64, TOK_ALL / 64), 256, 0, stream>>>(
      Mh, D_MODEL, Wfc, D_INNER, fc_b, nullptr, nullptr, Hh,
      TOK_ALL, D_INNER, D_MODEL, 1);

  // 8) MLP proj + residual -> y0 / y1 (fp32, straight into d_out)
  for (int e = 0; e < 2; ++e) {
    gemm_wmma_kernel<<<dim3(D_MODEL / 64, TOK_STREAM / 64), 256, 0, stream>>>(
        Hh + (size_t)e * TOK_STREAM * D_INNER, D_INNER,
        Wmproj, D_MODEL, mproj_b,
        Xmid + (size_t)e * TOK_STREAM * D_MODEL,
        outp + (size_t)e * Y_ELEMS, nullptr,
        TOK_STREAM, D_MODEL, D_INNER, 2);
  }

  // 9) pose head on y1: convert to f16 (reuse Ah), fc+GELU, tiny proj+normalize
  cvt(outp + Y_ELEMS, Ah, Y_ELEMS);
  gemm_wmma_kernel<<<dim3((2 * D_MODEL) / 64, TOK_STREAM / 64), 256, 0, stream>>>(
      Ah, D_MODEL, Wpfc, 2 * D_MODEL, pfc_b, nullptr, nullptr, PoseH,
      TOK_STREAM, 2 * D_MODEL, D_MODEL, 1);
  pose_proj_kernel<<<TOK_STREAM, 224, 0, stream>>>(
      PoseH, ppr_w, ppr_b, outp + 2 * (size_t)Y_ELEMS);
}
